// LSTMSenderReinforce_87179246174271
// MI455X (gfx1250) — compile-verified, matrix operand order checked by reading, not run on hardware
//
#include <hip/hip_runtime.h>
#include <hip/hip_bf16.h>
#include <math.h>

// Problem dims (fixed by the reference)
#define BB 256
#define TT 16
#define VV 32000
#define EE 256
#define HH 512
#define H4 2048

typedef __attribute__((ext_vector_type(16))) __bf16 v16bf;
typedef __attribute__((ext_vector_type(8)))  __bf16 v8bf;
typedef __attribute__((ext_vector_type(8)))  float  v8f;

// ---------------- CDNA5 async copy helpers ----------------
// Copy 64 contiguous bytes global -> LDS via the async-to-LDS path (ASYNCcnt).
// INST_OFFSET is applied to BOTH the LDS and global address (ISA 08 §4.4), so
// four b128 ops with offset 0/16/32/48 move 64B on both sides.
__device__ __forceinline__ void async_copy64(const void* gptr, void* lptr) {
  unsigned lds  = (unsigned)(uintptr_t)lptr;          // low 32b of generic = LDS offset
  unsigned long long ga = (unsigned long long)(uintptr_t)gptr;
  asm volatile(
      "global_load_async_to_lds_b128 %0, %1, off\n\t"
      "global_load_async_to_lds_b128 %0, %1, off offset:16\n\t"
      "global_load_async_to_lds_b128 %0, %1, off offset:32\n\t"
      "global_load_async_to_lds_b128 %0, %1, off offset:48"
      :: "v"(lds), "v"(ga)
      : "memory");
}
__device__ __forceinline__ void wait_async0() {
  asm volatile("s_wait_asynccnt 0x0" ::: "memory");
}

// ---------------- WMMA GEMM:  C[M,N] (+)= A[M,K] * W[N,K]^T + bias ----------------
// A: f32 row-major (activations, converted to bf16 on the LDS-staging path)
// W: bf16 row-major [N,K] (pre-converted weights), staged via async-to-LDS.
// Block = 256 threads = 8 waves (wave32). Tile: 64(M) x 256(N), K-step 32.
// Double-buffered LDS: next K-tile's async loads overlap current WMMAs.
#define MT 64
#define NT 256
#define KT 32

__global__ __launch_bounds__(256) void wmma_gemm_bt(
    const float* __restrict__ A, const __bf16* __restrict__ W,
    const float* __restrict__ bias, float* __restrict__ C,
    int M, int N, int K, int accumulate)
{
  __shared__ __align__(16) __bf16 As[2][MT][KT];   // 8 KB
  __shared__ __align__(16) __bf16 Bs[2][NT][KT];   // 32 KB
  const int tid   = threadIdx.x;
  const int lane  = tid & 31;
  const int wave  = tid >> 5;
  const int wm    = wave >> 2;        // 0..1
  const int wn    = wave & 3;         // 0..3
  const int lmod  = lane & 15;
  const int lhalf = lane >> 4;        // 0: K 0-7/16-23, 1: K 8-15/24-31
  const int blockM = blockIdx.y * MT;
  const int blockN = blockIdx.x * NT;

  v8f acc[2][4];
#pragma unroll
  for (int i = 0; i < 2; i++)
#pragma unroll
    for (int j = 0; j < 4; j++)
#pragma unroll
      for (int r = 0; r < 8; r++) acc[i][j][r] = 0.0f;

  const int arow = tid >> 2;          // 0..63
  const int aseg = (tid & 3) * 8;     // 0,8,16,24
  const int nk   = K / KT;

  auto stage = [&](int buf, int k0) {
    // W tile (256 rows x 32 bf16 = 64B/row): async global->LDS, one row/thread.
    async_copy64(W + (size_t)(blockN + tid) * K + k0, &Bs[buf][tid][0]);
    // A tile (64x32 f32 -> bf16): one 32B f32 chunk per thread via VGPRs.
    const float* s = A + (size_t)(blockM + arow) * K + (k0 + aseg);
    float4 f0 = ((const float4*)s)[0];
    float4 f1 = ((const float4*)s)[1];
    v8bf hbuf;
    hbuf[0] = (__bf16)f0.x; hbuf[1] = (__bf16)f0.y;
    hbuf[2] = (__bf16)f0.z; hbuf[3] = (__bf16)f0.w;
    hbuf[4] = (__bf16)f1.x; hbuf[5] = (__bf16)f1.y;
    hbuf[6] = (__bf16)f1.z; hbuf[7] = (__bf16)f1.w;
    *(v8bf*)&As[buf][arow][aseg] = hbuf;
  };

  // Prologue: stage tile 0.
  stage(0, 0);
  wait_async0();
  __syncthreads();

  for (int kt = 0; kt < nk; kt++) {
    const int cur = kt & 1;
    // Prefetch next K-tile into the other buffer (overlaps with WMMAs below;
    // the barrier at the end of the previous iteration guarantees nobody is
    // still reading that buffer).
    if (kt + 1 < nk) stage(cur ^ 1, (kt + 1) * KT);

    // Build fragments per CDNA5 16-bit operand layout.
    v16bf afrag[2], bfrag[4];
#pragma unroll
    for (int mi = 0; mi < 2; mi++) {
      int r = wm * 32 + mi * 16 + lmod;
      v8bf lo = *(const v8bf*)&As[cur][r][lhalf * 8];
      v8bf hi = *(const v8bf*)&As[cur][r][16 + lhalf * 8];
#pragma unroll
      for (int e = 0; e < 8; e++) { afrag[mi][e] = lo[e]; afrag[mi][e + 8] = hi[e]; }
    }
#pragma unroll
    for (int ni = 0; ni < 4; ni++) {
      int c = wn * 64 + ni * 16 + lmod;
      v8bf lo = *(const v8bf*)&Bs[cur][c][lhalf * 8];
      v8bf hi = *(const v8bf*)&Bs[cur][c][16 + lhalf * 8];
#pragma unroll
      for (int e = 0; e < 8; e++) { bfrag[ni][e] = lo[e]; bfrag[ni][e + 8] = hi[e]; }
    }
#pragma unroll
    for (int mi = 0; mi < 2; mi++)
#pragma unroll
      for (int ni = 0; ni < 4; ni++)
        acc[mi][ni] = __builtin_amdgcn_wmma_f32_16x16x32_bf16(
            false, afrag[mi], false, bfrag[ni], (short)0, acc[mi][ni], false, false);

    // Our async loads for the next tile must land before the barrier releases
    // everyone into reading that buffer.
    wait_async0();
    __syncthreads();
  }

  // Epilogue. D layout: VGPR r -> M = r + (lane<16 ? 0 : 8), N = lane%16.
#pragma unroll
  for (int mi = 0; mi < 2; mi++) {
    int baseM = blockM + wm * 32 + mi * 16 + lhalf * 8;
#pragma unroll
    for (int ni = 0; ni < 4; ni++) {
      int cn = blockN + wn * 64 + ni * 16 + lmod;
      float bv = bias ? bias[cn] : 0.0f;
#pragma unroll
      for (int r = 0; r < 8; r++) {
        size_t idx = (size_t)(baseM + r) * N + cn;
        float val = acc[mi][ni][r] + bv;
        C[idx] = accumulate ? (C[idx] + val) : val;
      }
    }
  }
}

// ---------------- Elementwise / reduction kernels ----------------
__global__ void cvt_f32_to_bf16(const float* __restrict__ src, __bf16* __restrict__ dst, int n) {
  int i = blockIdx.x * blockDim.x + threadIdx.x;
  int stride = gridDim.x * blockDim.x;
  for (; i < n; i += stride) dst[i] = (__bf16)src[i];
}

__global__ void zero_f32(float* __restrict__ p, size_t n) {  // n % 4 == 0
  size_t n4 = n >> 2;
  float4* p4 = (float4*)p;
  size_t i = (size_t)blockIdx.x * blockDim.x + threadIdx.x;
  size_t stride = (size_t)gridDim.x * blockDim.x;
  float4 z = make_float4(0.f, 0.f, 0.f, 0.f);
  for (; i < n4; i += stride) p4[i] = z;
}

__global__ void zero_i32(int* p, int n) {
  int i = blockIdx.x * blockDim.x + threadIdx.x;
  if (i < n) p[i] = 0;
}

__global__ void bcast_sos(const float* __restrict__ sos, float* __restrict__ inp) {
  int i = blockIdx.x * blockDim.x + threadIdx.x;  // over BB*EE
  inp[i] = sos[i & (EE - 1)];
}

__device__ __forceinline__ float gelu_exact(float v) {
  return 0.5f * v * (1.0f + erff(v * 0.70710678118654752440f));
}

__global__ __launch_bounds__(256) void ln_gelu_kernel(
    const float* __restrict__ z, const float* __restrict__ g,
    const float* __restrict__ b, float* __restrict__ out)
{
  __shared__ float red[256];
  int row = blockIdx.x, tid = threadIdx.x;
  const float* zr = z + (size_t)row * HH;
  float v0 = zr[tid], v1 = zr[tid + 256];
  red[tid] = v0 + v1; __syncthreads();
  for (int s = 128; s > 0; s >>= 1) { if (tid < s) red[tid] += red[tid + s]; __syncthreads(); }
  float mu = red[0] * (1.0f / HH);
  __syncthreads();
  float d0 = v0 - mu, d1 = v1 - mu;
  red[tid] = d0 * d0 + d1 * d1; __syncthreads();
  for (int s = 128; s > 0; s >>= 1) { if (tid < s) red[tid] += red[tid + s]; __syncthreads(); }
  float inv = rsqrtf(red[0] * (1.0f / HH) + 1e-5f);
  out[(size_t)row * HH + tid]       = gelu_exact(d0 * inv * g[tid]       + b[tid]);
  out[(size_t)row * HH + tid + 256] = gelu_exact(d1 * inv * g[tid + 256] + b[tid + 256]);
}

__global__ void lstm_pointwise(const float* __restrict__ gates,
                               float* __restrict__ h, float* __restrict__ c)
{
  int idx = blockIdx.x * blockDim.x + threadIdx.x;  // over BB*HH
  int b = idx >> 9, j = idx & (HH - 1);
  const float* g = gates + (size_t)b * H4;
  float ii = g[j], ff = g[HH + j], gg = g[2 * HH + j], oo = g[3 * HH + j];
  float si = 1.0f / (1.0f + __expf(-ii));
  float sf = 1.0f / (1.0f + __expf(-ff));
  float so = 1.0f / (1.0f + __expf(-oo));
  float cn = sf * c[idx] + si * tanhf(gg);
  c[idx] = cn;
  h[idx] = so * tanhf(cn);
}

__device__ __forceinline__ unsigned pcg(unsigned x) {
  x = x * 747796405u + 2891336453u;
  unsigned w = ((x >> ((x >> 28) + 4u)) ^ x) * 277803737u;
  return (w >> 22) ^ w;
}

// Block-per-row: Gumbel-max sampling + log-softmax + entropy over V=32000.
__global__ __launch_bounds__(256) void sample_kernel(
    const float* __restrict__ logits,
    float* __restrict__ o_ids, float* __restrict__ o_logp, float* __restrict__ o_ent,
    int* __restrict__ tok, int* __restrict__ done, int t, unsigned seed)
{
  __shared__ float sm[256];
  __shared__ float sg[256];
  __shared__ int   si[256];
  __shared__ float s_m;
  int b = blockIdx.x, tid = threadIdx.x;
  const float* l = logits + (size_t)b * VV;

  // Pass 1: row max + Gumbel-perturbed argmax.
  float m = -3.4e38f, gb = -3.4e38f; int gi = 0;
  for (int v = tid; v < VV; v += 256) {
    float lv = l[v];
    m = fmaxf(m, lv);
    unsigned h = pcg((unsigned)v ^ ((unsigned)b * 0x632be5abu) ^
                     ((unsigned)t * 0x9e3779b9u) ^ seed);
    float u = (h >> 8) * (1.0f / 16777216.0f);
    u = fminf(fmaxf(u, 1e-10f), 1.0f - 1e-7f);
    float s = lv - logf(-logf(u));
    if (s > gb) { gb = s; gi = v; }
  }
  sm[tid] = m; sg[tid] = gb; si[tid] = gi; __syncthreads();
  for (int s = 128; s > 0; s >>= 1) {
    if (tid < s) {
      sm[tid] = fmaxf(sm[tid], sm[tid + s]);
      if (sg[tid + s] > sg[tid]) { sg[tid] = sg[tid + s]; si[tid] = si[tid + s]; }
    }
    __syncthreads();
  }
  if (tid == 0) s_m = sm[0];
  __syncthreads();
  float mm = s_m;

  // Pass 2 (L2-resident re-read): S = sum e^(l-m), W = sum e^(l-m)*(l-m).
  float S = 0.0f, Wm = 0.0f;
  for (int v = tid; v < VV; v += 256) {
    float d = l[v] - mm;
    float e = __expf(d);
    S += e; Wm += e * d;
  }
  sm[tid] = S; sg[tid] = Wm; __syncthreads();
  for (int s = 128; s > 0; s >>= 1) {
    if (tid < s) { sm[tid] += sm[tid + s]; sg[tid] += sg[tid + s]; }
    __syncthreads();
  }
  if (tid == 0) {
    float St = sm[0], Wt = sg[0];
    float logS = logf(St);
    int d = done[b];
    int tk = d ? 0 : si[0];
    float lp = d ? 0.0f : (l[tk] - (mm + logS));
    float en = d ? 0.0f : (logS - Wt / St);
    o_ids[b * TT + t]  = (float)tk;
    o_logp[b * TT + t] = lp;
    o_ent[b * TT + t]  = en;
    tok[b] = tk;
    if (tk == 0) done[b] = 1;
  }
}

__global__ void gather_emb(const float* __restrict__ emb, const int* __restrict__ tok,
                           float* __restrict__ inp) {
  int i = blockIdx.x * blockDim.x + threadIdx.x;  // over BB*EE
  int b = i >> 8, e = i & (EE - 1);
  inp[i] = emb[(size_t)tok[b] * EE + e];
}

__global__ void finalize_kernel(float* __restrict__ ids, float* __restrict__ len) {
  int b = threadIdx.x;
  if (b >= BB) return;
  int first = TT;
  for (int t = 0; t < TT; t++) { if (ids[b * TT + t] == 0.0f) { first = t; break; } }
  int L = (first < TT) ? first + 1 : TT;
  len[b] = (float)L;
  for (int t = L - 1; t < TT; t++) ids[b * TT + t] = 0.0f;
}

__global__ void onehot_kernel(const float* __restrict__ ids, float* __restrict__ msg) {
  int i = blockIdx.x * blockDim.x + threadIdx.x;  // over BB*TT
  if (i < BB * TT) {
    int id = (int)ids[i];
    msg[(size_t)i * VV + id] = 1.0f;
  }
}

// ---------------- Host orchestration ----------------
extern "C" void kernel_launch(void* const* d_in, const int* in_sizes, int n_in,
                              void* d_out, int out_size, void* d_ws, size_t ws_size,
                              hipStream_t stream) {
  (void)in_sizes; (void)n_in; (void)out_size; (void)ws_size;
  const float* x     = (const float*)d_in[0];
  const float* bb_w  = (const float*)d_in[1];
  const float* bb_b  = (const float*)d_in[2];
  const float* ln_g  = (const float*)d_in[3];
  const float* ln_b  = (const float*)d_in[4];
  const float* emb   = (const float*)d_in[5];
  const float* sos   = (const float*)d_in[6];
  const float* wih0  = (const float*)d_in[7];
  const float* whh0  = (const float*)d_in[8];
  const float* bih0  = (const float*)d_in[9];
  const float* bhh0  = (const float*)d_in[10];
  const float* wih1  = (const float*)d_in[11];
  const float* whh1  = (const float*)d_in[12];
  const float* bih1  = (const float*)d_in[13];
  const float* bhh1  = (const float*)d_in[14];
  const float* out_w = (const float*)d_in[15];
  const float* out_b = (const float*)d_in[16];

  char* ws = (char*)d_ws;
  size_t off = 0;
  auto alloc = [&](size_t bytes) -> void* {
    void* p = ws + off; off += (bytes + 255) & ~(size_t)255; return p;
  };
  __bf16* bbw_h  = (__bf16*)alloc((size_t)HH * EE * 2);
  __bf16* wih0_h = (__bf16*)alloc((size_t)H4 * EE * 2);
  __bf16* whh0_h = (__bf16*)alloc((size_t)H4 * HH * 2);
  __bf16* wih1_h = (__bf16*)alloc((size_t)H4 * HH * 2);
  __bf16* whh1_h = (__bf16*)alloc((size_t)H4 * HH * 2);
  __bf16* outw_h = (__bf16*)alloc((size_t)VV * HH * 2);
  float* zbuf   = (float*)alloc((size_t)BB * HH * 4);
  float* h0buf  = (float*)alloc((size_t)BB * HH * 4);
  float* c0buf  = (float*)alloc((size_t)BB * HH * 4);
  float* h1buf  = (float*)alloc((size_t)BB * HH * 4);
  float* c1buf  = (float*)alloc((size_t)BB * HH * 4);
  float* inpbuf = (float*)alloc((size_t)BB * EE * 4);
  float* gates  = (float*)alloc((size_t)BB * H4 * 4);
  float* logits = (float*)alloc((size_t)BB * VV * 4);
  int* tok  = (int*)alloc(BB * 4);
  int* done = (int*)alloc(BB * 4);

  float* out    = (float*)d_out;
  float* o_ids  = out;
  float* o_msg  = out + BB * TT;
  float* o_logp = o_msg + (size_t)BB * TT * VV;
  float* o_ent  = o_logp + BB * TT;
  float* o_len  = o_ent + BB * TT;

  // Weights -> bf16 (done every call; deterministic).
  cvt_f32_to_bf16<<<512, 256, 0, stream>>>(bb_w,  bbw_h,  HH * EE);
  cvt_f32_to_bf16<<<1024, 256, 0, stream>>>(wih0, wih0_h, H4 * EE);
  cvt_f32_to_bf16<<<2048, 256, 0, stream>>>(whh0, whh0_h, H4 * HH);
  cvt_f32_to_bf16<<<2048, 256, 0, stream>>>(wih1, wih1_h, H4 * HH);
  cvt_f32_to_bf16<<<2048, 256, 0, stream>>>(whh1, whh1_h, H4 * HH);
  cvt_f32_to_bf16<<<4096, 256, 0, stream>>>(out_w, outw_h, VV * HH);

  // State init.
  zero_f32<<<128, 256, 0, stream>>>(c0buf, (size_t)BB * HH);
  zero_f32<<<128, 256, 0, stream>>>(h1buf, (size_t)BB * HH);
  zero_f32<<<128, 256, 0, stream>>>(c1buf, (size_t)BB * HH);
  zero_i32<<<1, 256, 0, stream>>>(done, BB);
  bcast_sos<<<(BB * EE) / 256, 256, 0, stream>>>(sos, inpbuf);

  // Backbone: z = x@bb_w^T + bb_b ; h0 = GELU(LN(z)).
  wmma_gemm_bt<<<dim3(HH / NT, BB / MT), 256, 0, stream>>>(x, bbw_h, bb_b, zbuf, BB, HH, EE, 0);
  ln_gelu_kernel<<<BB, 256, 0, stream>>>(zbuf, ln_g, ln_b, h0buf);

  // Zero the one-hot msg output (524 MB).
  zero_f32<<<8192, 256, 0, stream>>>(o_msg, (size_t)BB * TT * VV);

  for (int t = 0; t < TT; t++) {
    // Layer 0: gates = inp@wih0^T + bih0 + h0@whh0^T + bhh0
    wmma_gemm_bt<<<dim3(H4 / NT, BB / MT), 256, 0, stream>>>(inpbuf, wih0_h, bih0, gates, BB, H4, EE, 0);
    wmma_gemm_bt<<<dim3(H4 / NT, BB / MT), 256, 0, stream>>>(h0buf,  whh0_h, bhh0, gates, BB, H4, HH, 1);
    lstm_pointwise<<<(BB * HH) / 256, 256, 0, stream>>>(gates, h0buf, c0buf);
    // Layer 1
    wmma_gemm_bt<<<dim3(H4 / NT, BB / MT), 256, 0, stream>>>(h0buf, wih1_h, bih1, gates, BB, H4, HH, 0);
    wmma_gemm_bt<<<dim3(H4 / NT, BB / MT), 256, 0, stream>>>(h1buf, whh1_h, bhh1, gates, BB, H4, HH, 1);
    lstm_pointwise<<<(BB * HH) / 256, 256, 0, stream>>>(gates, h1buf, c1buf);
    // Logits: [256 x 32000], K=512 — the dominant WMMA GEMM.
    wmma_gemm_bt<<<dim3(VV / NT, BB / MT), 256, 0, stream>>>(h1buf, outw_h, out_b, logits, BB, VV, HH, 0);
    // Sample + logp + entropy + done-masking.
    sample_kernel<<<BB, 256, 0, stream>>>(logits, o_ids, o_logp, o_ent, tok, done, t, 0xC0FFEEu);
    // Next input = emb[tok].
    gather_emb<<<(BB * EE) / 256, 256, 0, stream>>>(emb, tok, inpbuf);
  }

  finalize_kernel<<<1, 256, 0, stream>>>(o_ids, o_len);
  onehot_kernel<<<(BB * TT) / 256, 256, 0, stream>>>(o_ids, o_msg);
}
